// ContrastiveLoss_45277545235064
// MI455X (gfx1250) — matrix operand order repaired and checked by reading, not compiled, not run on hardware
//
#include <hip/hip_runtime.h>

typedef __attribute__((ext_vector_type(16))) _Float16 v16h;
typedef __attribute__((ext_vector_type(8)))  _Float16 v8h;
typedef __attribute__((ext_vector_type(8)))  float    v8f;

#define N_TOT   8192
#define DIM     256
#define INV_TAU 10.0f
#define SELF_E  22026.465794806718f   /* exp(1/tau) = exp(10) */

// ---------------------------------------------------------------------------
// Kernel 1: row-normalize f32 -> f16, one wave32 per row; also zero pos/neg.
// ---------------------------------------------------------------------------
__global__ __launch_bounds__(256) void cl_normalize(const float* __restrict__ f,
                                                    _Float16* __restrict__ fh,
                                                    float* __restrict__ pos,
                                                    float* __restrict__ neg) {
  int gt = blockIdx.x * 256 + threadIdx.x;
  if (gt < N_TOT) { pos[gt] = 0.f; neg[gt] = 0.f; }

  int row  = gt >> 5;          // one wave (32 lanes) per row
  int lane = threadIdx.x & 31; // 8 floats per lane (256/32)
  const float4* rp = (const float4*)(f + (size_t)row * DIM + lane * 8);
  float4 a = rp[0], b = rp[1];
  float ss = a.x*a.x + a.y*a.y + a.z*a.z + a.w*a.w
           + b.x*b.x + b.y*b.y + b.z*b.z + b.w*b.w;
  #pragma unroll
  for (int m = 16; m >= 1; m >>= 1) ss += __shfl_xor(ss, m, 32);
  float inv = rsqrtf(ss);
  inv = inv * (1.5f - 0.5f * ss * inv * inv);   // one Newton step

  v8h o;
  o[0] = (_Float16)(a.x*inv); o[1] = (_Float16)(a.y*inv);
  o[2] = (_Float16)(a.z*inv); o[3] = (_Float16)(a.w*inv);
  o[4] = (_Float16)(b.x*inv); o[5] = (_Float16)(b.y*inv);
  o[6] = (_Float16)(b.z*inv); o[7] = (_Float16)(b.w*inv);
  *(v8h*)(fh + (size_t)row * DIM + lane * 8) = o;
}

// ---------------------------------------------------------------------------
// Kernel 2: fused 128x128-tile WMMA GEMM + exp + masked row reduction.
//   grid = (64, 64): blockIdx.x = col tile, blockIdx.y = row tile.
//   256 threads = 8 waves; wave w owns rows [rowBase + 16w, +16).
//   B tile staged with GLOBAL_LOAD_ASYNC_TO_LDS_B128 (ASYNCcnt path, no VGPR
//   round-trip) into 64 KiB LDS, K-chunk-major:
//     bt[kchunk(0..31)][col(0..127)][8 halfs]
//   -> 16-lane B fragment reads are bank-conflict-free without padding.
// ---------------------------------------------------------------------------
__global__ __launch_bounds__(256) void cl_tile(const _Float16* __restrict__ fh,
                                               float* __restrict__ pos,
                                               float* __restrict__ neg) {
  __shared__ _Float16 bt[32 * 1024];   // 32 kchunks * 128 cols * 8 halfs = 64 KiB
  const int colBase = blockIdx.x * 128;
  const int rowBase = blockIdx.y * 128;
  const int t = threadIdx.x;

  // ---- async-stage B tile (cols colBase..+127, all 256 K) into LDS ----
  {
    const int col = t & 127, seg = t >> 7;   // 2 threads per column
    const char* src = (const char*)(fh + (size_t)(colBase + col) * DIM + seg * 128);
    // low 32 bits of a generic pointer into __shared__ == wave-relative LDS offset
    const unsigned ldsBase =
        (unsigned)(size_t)((const char*)bt) + (unsigned)((seg * 16) * 128 + col) * 16u;
    #pragma unroll
    for (int j = 0; j < 16; ++j) {
      unsigned ldsOff = ldsBase + (unsigned)(j * 128 * 16);   // kchunk = seg*16+j
      asm volatile("global_load_async_to_lds_b128 %0, %1, off"
                   :: "v"(ldsOff), "v"(src + j * 16) : "memory");
    }
    asm volatile("s_wait_asynccnt 0" ::: "memory");
  }
  __syncthreads();

  const int wave = t >> 5, lane = t & 31;
  const int hl = lane >> 4, l15 = lane & 15;

  // A fragment source: row = lane&15 within the wave's 16-row strip;
  // half-wave selects K sub-block per ISA 16-bit A layout.
  const int arow = rowBase + wave * 16 + l15;
  const _Float16* aptr = fh + (size_t)arow * DIM + hl * 8;

  v8f acc[8] = {};   // 8 col-subtiles of 16x16 f32

  #pragma unroll
  for (int ks = 0; ks < 8; ++ks) {            // K = 256 in steps of 32
    v8h alo = *(const v8h*)(aptr + ks * 32);        // K kb..kb+7
    v8h ahi = *(const v8h*)(aptr + ks * 32 + 16);   // K kb+16..kb+23
    v16h amat = __builtin_shufflevector(alo, ahi,
        0,1,2,3,4,5,6,7,8,9,10,11,12,13,14,15);
    #pragma unroll
    for (int ct = 0; ct < 8; ++ct) {
      // B fragment: col = ct*16 + (lane&15); K = ks*32 + hl*8 .. +7, ks*32 + 16 + hl*8 .. +7
      const _Float16* bp = bt + (ks * 4 + hl) * 1024 + (ct * 16 + l15) * 8;
      v8h blo = *(const v8h*)bp;            // kchunk ks*4+hl
      v8h bhi = *(const v8h*)(bp + 2048);   // kchunk ks*4+2+hl
      v16h bmat = __builtin_shufflevector(blo, bhi,
          0,1,2,3,4,5,6,7,8,9,10,11,12,13,14,15);
      acc[ct] = __builtin_amdgcn_wmma_f32_16x16x32_f16(
          /*neg_a=*/false, amat, /*neg_b=*/false, bmat,
          /*c_mod=*/(short)0, acc[ct], /*reuse_a=*/false, /*reuse_b=*/false);
    }
  }

  // ---- epilogue: E = exp(S/tau), split into same-group / other-group sums ----
  // C/D layout: VGPR v, lanes 0-15 -> row v,   col = lane
  //                     lanes 16-31 -> row v+8, col = lane-16
  float rp_[8], rn_[8];
  #pragma unroll
  for (int v = 0; v < 8; ++v) { rp_[v] = 0.f; rn_[v] = 0.f; }

  const int crowBase = rowBase + wave * 16 + hl * 8;
  #pragma unroll
  for (int ct = 0; ct < 8; ++ct) {
    const int cg = (colBase + ct * 16 + l15) >> 3;   // group of this lane's column
    #pragma unroll
    for (int v = 0; v < 8; ++v) {
      float e = __expf(acc[ct][v] * INV_TAU);
      bool same = (((crowBase + v) >> 3) == cg);     // groups = blocks of 8 rows
      rp_[v] += same ? e : 0.f;
      rn_[v] += same ? 0.f : e;
    }
  }

  // butterfly reduce across the 16 lanes that share each row, then atomics
  #pragma unroll
  for (int v = 0; v < 8; ++v) {
    float p = rp_[v], n = rn_[v];
    #pragma unroll
    for (int m = 8; m >= 1; m >>= 1) {   // masks 1..8 stay within 16-lane half
      p += __shfl_xor(p, m, 32);
      n += __shfl_xor(n, m, 32);
    }
    if (l15 == 0) {
      atomicAdd(&pos[crowBase + v], p);
      atomicAdd(&neg[crowBase + v], n);
    }
  }
}

// ---------------------------------------------------------------------------
// Kernel 3: loss = sum(-log((pos - e^{1/tau}) / neg)) / N
// ---------------------------------------------------------------------------
__global__ __launch_bounds__(256) void cl_loss(const float* __restrict__ pos,
                                               const float* __restrict__ neg,
                                               float* __restrict__ out) {
  __shared__ float red[256];
  int t = threadIdx.x;
  float s = 0.f;
  for (int r = t; r < N_TOT; r += 256)
    s += -__logf((pos[r] - SELF_E) / neg[r]);
  red[t] = s;
  __syncthreads();
  #pragma unroll
  for (int k = 128; k > 0; k >>= 1) {
    if (t < k) red[t] += red[t + k];
    __syncthreads();
  }
  if (t == 0) out[0] = red[0] / (float)N_TOT;
}

// ---------------------------------------------------------------------------
extern "C" void kernel_launch(void* const* d_in, const int* in_sizes, int n_in,
                              void* d_out, int out_size, void* d_ws, size_t ws_size,
                              hipStream_t stream) {
  const float* features = (const float*)d_in[0];
  // d_in[1] = num_crops (int64[1024], constant 8 per group) -> group id = row >> 3.
  (void)in_sizes; (void)n_in; (void)ws_size; (void)out_size;

  _Float16* fh = (_Float16*)d_ws;                             // 8192*256*2 = 4 MiB
  float* pos   = (float*)((char*)d_ws + (size_t)N_TOT * DIM * 2);
  float* neg   = pos + N_TOT;

  // 8192 rows, one wave per row -> 8192*32 threads / 256 = 1024 blocks
  cl_normalize<<<1024, 256, 0, stream>>>(features, fh, pos, neg);

  dim3 grid(64, 64);   // 128x128 output tiles over 8192x8192
  cl_tile<<<grid, 256, 0, stream>>>(fh, pos, neg);

  cl_loss<<<1, 256, 0, stream>>>(pos, neg, (float*)d_out);
}
